// GPTClosedLoop_64613488001449
// MI455X (gfx1250) — matrix operand order repaired and checked by reading, not compiled, not run on hardware
//
#include <hip/hip_runtime.h>
#include <hip/hip_bf16.h>
#include <stdint.h>

// ---------------------------------------------------------------------------
// GPT closed-loop (B=32, T=129, L=6, D=128, H=4, DH=32) on MI455X / gfx1250.
// Strategy: exact KV-cache incremental decode, one persistent workgroup
// (256 threads = 8 wave32s) runs all 128 sequential steps; activations in
// LDS, weights prepacked to f16 WMMA-B fragment layout in workspace,
// matmuls via v_wmma_f32_16x16x32_f16.
// ---------------------------------------------------------------------------

typedef __attribute__((ext_vector_type(16))) _Float16 v16h;
typedef __attribute__((ext_vector_type(8)))  _Float16 v8h;
typedef __attribute__((ext_vector_type(8)))  float    v8f;

#define GB   32      // batch
#define GT   129     // T
#define GTH  128     // T_H (steps)
#define GL   6       // layers
#define GD   128     // model dim
#define GH   4       // heads
#define GDH  32      // head dim
#define NTHR 256
#define NWAV 8

// ---- WMMA helpers (layouts per CDNA5 ISA 7.12.2) --------------------------

__device__ __forceinline__ v16h load_a_frag(const _Float16* A, int rowBase,
                                            int lda, int kt, int lane) {
  const int m    = lane & 15;
  const int half = lane >> 4;
  const _Float16* p = A + (size_t)(rowBase + m) * lda + kt * 32 + half * 8;
  union { v16h v; v8h h[2]; } u;
  u.h[0] = *reinterpret_cast<const v8h*>(p);        // K = kt*32 + 8*half + 0..7
  u.h[1] = *reinterpret_cast<const v8h*>(p + 16);   // K = kt*32 + 16 + 8*half + 0..7
  return u.v;
}

__device__ __forceinline__ v16h load_b_frag(const _Float16* Bpack, int tileIdx,
                                            int lane) {
  const _Float16* p = Bpack + ((size_t)tileIdx * 32 + lane) * 16;  // 32B/lane
  union { v16h v; v8h h[2]; } u;
  u.h[0] = *reinterpret_cast<const v8h*>(p);
  u.h[1] = *reinterpret_cast<const v8h*>(p + 8);
  return u.v;
}

__device__ __forceinline__ v8f wmma_acc(v16h a, v16h b, v8f c) {
  return __builtin_amdgcn_wmma_f32_16x16x32_f16(false, a, false, b,
                                                (short)0, c, false, false);
}

// ---- misc device helpers --------------------------------------------------

__device__ __forceinline__ float wave_reduce_add(float v) {
  for (int off = 16; off; off >>= 1) v += __shfl_xor(v, off, 32);
  return v;
}

__device__ __forceinline__ float gelu_tanh(float v) {
  const float k = 0.7978845608028654f;
  return 0.5f * v * (1.0f + tanhf(k * (v + 0.044715f * v * v * v)));
}

// LayerNorm rows of xs (32 x 128 f32) -> f16 out, one wave per token.
__device__ __forceinline__ void ln_to_f16(const float* xs, _Float16* out,
                                          const float* w, const float* b,
                                          int wave, int lane) {
  for (int tok = wave; tok < GB; tok += NWAV) {
    const float* row = xs + tok * GD;
    float v[4], sum = 0.f, sq = 0.f;
#pragma unroll
    for (int j = 0; j < 4; ++j) {
      v[j] = row[lane + 32 * j];
      sum += v[j];
      sq  += v[j] * v[j];
    }
    sum = wave_reduce_add(sum);
    sq  = wave_reduce_add(sq);
    const float mean = sum * (1.f / GD);
    const float var  = sq * (1.f / GD) - mean * mean;
    const float rstd = rsqrtf(var + 1e-5f);
#pragma unroll
    for (int j = 0; j < 4; ++j) {
      const int d = lane + 32 * j;
      out[tok * GD + d] = (_Float16)((v[j] - mean) * rstd * w[d] + b[d]);
    }
  }
}

// ---- prepack: fp32 (L,K,N) row-major -> f16 WMMA-B fragment order ---------
// dst[(((l*nnt+nt)*nkt+kt)*32+lane)*16 + j] = src[l][kt*32+(lane/16)*8+(j%8)+(j/8)*16][nt*16+lane%16]

__global__ void prepack_b_kernel(const float* __restrict__ src,
                                 _Float16* __restrict__ dst,
                                 int layers, int K, int N) {
  const int nkt = K / 32, nnt = N / 16;
  const int64_t total = (int64_t)layers * K * N;
  for (int64_t p = (int64_t)blockIdx.x * blockDim.x + threadIdx.x; p < total;
       p += (int64_t)gridDim.x * blockDim.x) {
    const int     j    = (int)(p & 15);
    const int     lane = (int)((p >> 4) & 31);
    const int64_t t    = p >> 9;                 // (l*nnt + nt)*nkt + kt
    const int     kt   = (int)(t % nkt);
    const int     nt   = (int)((t / nkt) % nnt);
    const int     l    = (int)(t / (nkt * (int64_t)nnt));
    const int row = kt * 32 + ((lane >> 4) << 3) + (j & 7) + ((j >> 3) << 4);
    const int col = nt * 16 + (lane & 15);
    dst[p] = (_Float16)src[((int64_t)l * K + row) * N + col];
  }
}

// ---- persistent closed-loop kernel ----------------------------------------

__global__ __launch_bounds__(NTHR)
void gpt_closed_loop_kernel(const float* __restrict__ data,
                            const float* __restrict__ r,
                            const float* __restrict__ wte_w,
                            const float* __restrict__ wte_b,
                            const float* __restrict__ wpe,
                            const float* __restrict__ ln1_w,
                            const float* __restrict__ ln1_b,
                            const float* __restrict__ ln2_w,
                            const float* __restrict__ ln2_b,
                            const float* __restrict__ qkv_b,
                            const float* __restrict__ attn_proj_b,
                            const float* __restrict__ fc_b,
                            const float* __restrict__ fc2_b,
                            const float* __restrict__ lnf_w,
                            const float* __restrict__ lnf_b,
                            const float* __restrict__ head_w,
                            const float* __restrict__ head_b,
                            const _Float16* __restrict__ qkv_pk,
                            const _Float16* __restrict__ proj_pk,
                            const _Float16* __restrict__ fc_pk,
                            const _Float16* __restrict__ fc2_pk,
                            float* __restrict__ kcache,
                            float* __restrict__ vcache,
                            float* __restrict__ Y) {
  extern __shared__ char smem[];
  float*    xs  = (float*)smem;                          // 32x128 f32  (16 KB)
  _Float16* hs  = (_Float16*)(smem + 16384);             // 32x128 f16  ( 8 KB)
  float*    qs  = (float*)(smem + 16384 + 8192);         // 32x384 f32  (48 KB)
  _Float16* fca = (_Float16*)qs;                         // alias: 32x512 f16
  float*    sst  = (float*)(smem + 16384 + 8192 + 49152);
  float*    ucur = sst + GB;
  float*    evec = ucur + GB;
  float*    unew = evec + GB;

  const int tid  = threadIdx.x;
  const int lane = tid & 31;
  const int wave = tid >> 5;
  const float scale = 0.17677669529663687f;  // 1/sqrt(32)

  if (tid < GB) { sst[tid] = 0.f; ucur[tid] = 0.f; }
  __syncthreads();

  for (int i = 0; i < GTH; ++i) {
    // -- scan-step prologue: e_i = r[:,i] - s ; Y[:,i] = s -------------------
    if (tid < GB) {
      const float s = sst[tid];
      evec[tid] = r[tid * GTH + i] - s;
      Y[tid * GTH + i] = s;
    }
    __syncthreads();

    // -- embedding for the new token at position i --------------------------
    for (int p = tid; p < GB * GD; p += NTHR) {
      const int b = p >> 7, d = p & 127;
      xs[p] = evec[b] * wte_w[d] + ucur[b] * wte_w[GD + d] + wte_b[d] +
              wpe[i * GD + d];
    }
    __syncthreads();

    for (int l = 0; l < GL; ++l) {
      // ---- LN1 -> hs (f16) ----
      ln_to_f16(xs, hs, ln1_w + l * GD, ln1_b + l * GD, wave, lane);
      __syncthreads();

      // ---- qkv = hs(32x128) @ W(128x384) + b -> qs f32 ----
      {
        const _Float16* Bp = qkv_pk + (size_t)l * GD * 384;
        const float* bias = qkv_b + l * 384;
        const int nkt = 4;  // K=128
        for (int tile = wave; tile < 48; tile += NWAV) {
          const int mt = tile & 1, nt = tile >> 1;
          const int col   = nt * 16 + (lane & 15);
          const int rbase = mt * 16 + ((lane >> 4) << 3);
          v8f c = {};
          const float bv = bias[col];
#pragma unroll
          for (int q = 0; q < 8; ++q) c[q] = bv;
#pragma unroll
          for (int kt = 0; kt < nkt; ++kt)
            c = wmma_acc(load_a_frag(hs, mt * 16, GD, kt, lane),
                         load_b_frag(Bp, nt * nkt + kt, lane), c);
#pragma unroll
          for (int q = 0; q < 8; ++q) qs[(rbase + q) * 384 + col] = c[q];
        }
      }
      __syncthreads();

      // ---- append K,V for position i into cache (L2-resident) ----
      for (int p = tid; p < GB * GD; p += NTHR) {
        const int b = p >> 7, hd = p & 127;
        const size_t idx =
            ((((size_t)l * GB + b) * GH + (hd >> 5)) * GT + i) * GDH +
            (hd & 31);
        kcache[idx] = qs[b * 384 + GD + hd];
        vcache[idx] = qs[b * 384 + 2 * GD + hd];
      }
      // prefetch next layer's fc weights into L2 while attention runs
      if (l + 1 < GL) {
        const char* nb = (const char*)(fc_pk + (size_t)(l + 1) * GD * 512);
        __builtin_prefetch(nb + (size_t)tid * 512, 0, 0);
      }
      __syncthreads();

      // ---- attention: one (b,h) pair per wave, lane = head-dim ----
      for (int pair = wave; pair < GB * GH; pair += NWAV) {
        const int b = pair >> 2, hh = pair & 3;
        const float qv = qs[b * 384 + hh * GDH + lane] * scale;
        const float* Kc =
            kcache + (((size_t)l * GB + b) * GH + hh) * (size_t)GT * GDH;
        const float* Vc =
            vcache + (((size_t)l * GB + b) * GH + hh) * (size_t)GT * GDH;
        float m = -1e30f, lsum = 0.f, acc = 0.f;
        for (int t = 0; t <= i; ++t) {
          float sc = wave_reduce_add(qv * Kc[t * GDH + lane]);
          const float nm   = fmaxf(m, sc);
          const float corr = __expf(m - nm);
          const float pe   = __expf(sc - nm);
          lsum = lsum * corr + pe;
          acc  = acc * corr + pe * Vc[t * GDH + lane];
          m = nm;
        }
        hs[b * GD + hh * GDH + lane] = (_Float16)(acc / lsum);
      }
      __syncthreads();

      // ---- x += attn_out(32x128) @ Wp(128x128) + b ----
      {
        const _Float16* Bp = proj_pk + (size_t)l * GD * GD;
        const float* bias = attn_proj_b + l * GD;
        for (int tile = wave; tile < 16; tile += NWAV) {
          const int mt = tile & 1, nt = tile >> 1;
          const int col   = nt * 16 + (lane & 15);
          const int rbase = mt * 16 + ((lane >> 4) << 3);
          v8f c = {};
          const float bv = bias[col];
#pragma unroll
          for (int q = 0; q < 8; ++q) c[q] = xs[(rbase + q) * GD + col] + bv;
#pragma unroll
          for (int kt = 0; kt < 4; ++kt)
            c = wmma_acc(load_a_frag(hs, mt * 16, GD, kt, lane),
                         load_b_frag(Bp, nt * 4 + kt, lane), c);
#pragma unroll
          for (int q = 0; q < 8; ++q) xs[(rbase + q) * GD + col] = c[q];
        }
      }
      __syncthreads();

      // ---- LN2 -> hs ----
      ln_to_f16(xs, hs, ln2_w + l * GD, ln2_b + l * GD, wave, lane);
      __syncthreads();

      // ---- fc: gelu(hs(32x128) @ W(128x512) + b) -> fca f16 ----
      {
        const _Float16* Bp = fc_pk + (size_t)l * GD * 512;
        const float* bias = fc_b + l * 512;
        for (int tile = wave; tile < 64; tile += NWAV) {
          const int mt = tile & 1, nt = tile >> 1;
          const int col   = nt * 16 + (lane & 15);
          const int rbase = mt * 16 + ((lane >> 4) << 3);
          v8f c = {};
          const float bv = bias[col];
#pragma unroll
          for (int q = 0; q < 8; ++q) c[q] = bv;
#pragma unroll
          for (int kt = 0; kt < 4; ++kt)
            c = wmma_acc(load_a_frag(hs, mt * 16, GD, kt, lane),
                         load_b_frag(Bp, nt * 4 + kt, lane), c);
#pragma unroll
          for (int q = 0; q < 8; ++q)
            fca[(rbase + q) * 512 + col] = (_Float16)gelu_tanh(c[q]);
        }
      }
      __syncthreads();

      // ---- x += fca(32x512) @ W2(512x128) + b ----
      {
        const _Float16* Bp = fc2_pk + (size_t)l * 512 * GD;
        const float* bias = fc2_b + l * GD;
        for (int tile = wave; tile < 16; tile += NWAV) {
          const int mt = tile & 1, nt = tile >> 1;
          const int col   = nt * 16 + (lane & 15);
          const int rbase = mt * 16 + ((lane >> 4) << 3);
          v8f c = {};
          const float bv = bias[col];
#pragma unroll
          for (int q = 0; q < 8; ++q) c[q] = xs[(rbase + q) * GD + col] + bv;
#pragma unroll
          for (int kt = 0; kt < 16; ++kt)
            c = wmma_acc(load_a_frag(fca, mt * 16, 512, kt, lane),
                         load_b_frag(Bp, nt * 16 + kt, lane), c);
#pragma unroll
          for (int q = 0; q < 8; ++q) xs[(rbase + q) * GD + col] = c[q];
        }
      }
      __syncthreads();
    }  // layers

    // ---- final LN + head (one wave per token) ----
    for (int tok = wave; tok < GB; tok += NWAV) {
      const float* row = xs + tok * GD;
      float v[4], sum = 0.f, sq = 0.f;
#pragma unroll
      for (int j = 0; j < 4; ++j) {
        v[j] = row[lane + 32 * j];
        sum += v[j];
        sq  += v[j] * v[j];
      }
      sum = wave_reduce_add(sum);
      sq  = wave_reduce_add(sq);
      const float mean = sum * (1.f / GD);
      const float rstd = rsqrtf(sq * (1.f / GD) - mean * mean + 1e-5f);
      float dot = 0.f;
#pragma unroll
      for (int j = 0; j < 4; ++j) {
        const int d = lane + 32 * j;
        dot += ((v[j] - mean) * rstd * lnf_w[d] + lnf_b[d]) * head_w[d];
      }
      dot = wave_reduce_add(dot);
      if (lane == 0) unew[tok] = dot + head_b[0];
    }
    __syncthreads();

    // ---- state update: s += -p0*s + p1*tanh(u_next) -----------------------
    if (tid < GB) {
      const float u  = unew[tid];
      const float s  = sst[tid];
      const float p0 = data[tid * 2 + 0];
      const float p1 = data[tid * 2 + 1];
      ucur[tid] = u;
      sst[tid]  = s + (-p0 * s + p1 * tanhf(u));
    }
    __syncthreads();
  }
}

// ---------------------------------------------------------------------------

extern "C" void kernel_launch(void* const* d_in, const int* in_sizes, int n_in,
                              void* d_out, int out_size, void* d_ws,
                              size_t ws_size, hipStream_t stream) {
  (void)in_sizes; (void)n_in; (void)out_size; (void)ws_size;

  const float* data        = (const float*)d_in[0];
  const float* r           = (const float*)d_in[1];
  const float* wte_w       = (const float*)d_in[2];
  const float* wte_b       = (const float*)d_in[3];
  const float* wpe         = (const float*)d_in[4];
  const float* ln1_w       = (const float*)d_in[5];
  const float* ln1_b       = (const float*)d_in[6];
  const float* ln2_w       = (const float*)d_in[7];
  const float* ln2_b       = (const float*)d_in[8];
  const float* qkv_w       = (const float*)d_in[9];
  const float* qkv_b       = (const float*)d_in[10];
  const float* attn_proj_w = (const float*)d_in[11];
  const float* attn_proj_b = (const float*)d_in[12];
  const float* fc_w        = (const float*)d_in[13];
  const float* fc_b        = (const float*)d_in[14];
  const float* fc2_w       = (const float*)d_in[15];
  const float* fc2_b       = (const float*)d_in[16];
  const float* lnf_w       = (const float*)d_in[17];
  const float* lnf_b       = (const float*)d_in[18];
  const float* head_w      = (const float*)d_in[19];
  const float* head_b      = (const float*)d_in[20];
  float* Y = (float*)d_out;

  // ---- workspace carve ----
  char* ws = (char*)d_ws;
  size_t off = 0;
  auto carve = [&](size_t bytes) -> void* {
    void* p = ws + off;
    off += (bytes + 255) & ~(size_t)255;
    return p;
  };
  _Float16* qkv_pk  = (_Float16*)carve((size_t)GL * GD * 384 * 2);
  _Float16* proj_pk = (_Float16*)carve((size_t)GL * GD * GD * 2);
  _Float16* fc_pk   = (_Float16*)carve((size_t)GL * GD * 512 * 2);
  _Float16* fc2_pk  = (_Float16*)carve((size_t)GL * 512 * GD * 2);
  float* kcache = (float*)carve((size_t)GL * GB * GH * GT * GDH * 4);
  float* vcache = (float*)carve((size_t)GL * GB * GH * GT * GDH * 4);

  // ---- prepack weights to f16 WMMA-B fragment layout (recomputed per call) --
  auto pk = [&](const float* src, _Float16* dst, int K, int N) {
    const int64_t total = (int64_t)GL * K * N;
    const int grid = (int)((total + NTHR - 1) / NTHR);
    prepack_b_kernel<<<grid, NTHR, 0, stream>>>(src, dst, GL, K, N);
  };
  pk(qkv_w,       qkv_pk,  GD,  384);
  pk(attn_proj_w, proj_pk, GD,  GD);
  pk(fc_w,        fc_pk,   GD,  512);
  pk(fc2_w,       fc2_pk,  512, GD);

  // ---- persistent single-WGP closed loop ----
  const size_t smem = 16384 + 8192 + 49152 + 4 * GB * sizeof(float);
  gpt_closed_loop_kernel<<<1, NTHR, smem, stream>>>(
      data, r, wte_w, wte_b, wpe, ln1_w, ln1_b, ln2_w, ln2_b, qkv_b,
      attn_proj_b, fc_b, fc2_b, lnf_w, lnf_b, head_w, head_b, qkv_pk, proj_pk,
      fc_pk, fc2_pk, kcache, vcache, Y);
}
